// CausalSelfAttentionBlock_69226282877407
// MI455X (gfx1250) — matrix operand order repaired and checked by reading, not compiled, not run on hardware
//
#include <hip/hip_runtime.h>
#include <hip/hip_bf16.h>
#include <math.h>

// ---------------------------------------------------------------------------
// Transformer block for MI455X (gfx1250, wave32, WMMA bf16 -> f32 accum)
// GEMM B-tiles staged to LDS via Tensor Data Mover (TENSORcnt double-buffer).
// ---------------------------------------------------------------------------

typedef __bf16 bf16_t;
typedef __attribute__((ext_vector_type(16))) __bf16 v16bf;
typedef __attribute__((ext_vector_type(8)))  __bf16 v8bf;
typedef __attribute__((ext_vector_type(8)))  float  v8f;
typedef __attribute__((ext_vector_type(4)))  unsigned int u32x4;
typedef __attribute__((ext_vector_type(8)))  int i32x8;
typedef __attribute__((ext_vector_type(4)))  int i32x4;

#define EMB     512
#define TSEQ    1024
#define BATCH   8
#define HEADS   8
#define HDIM    64
#define QKVN    1536
#define DFF     1365
#define DFFP    1408                 // padded to a multiple of 128 (TDM chunk)
#define MROWS   8192                 // BATCH*TSEQ
#define TOTALN  4194304              // MROWS*EMB
#define ATT_SCALE 0.04419417382415922f   // 1/sqrt(512)

// workspace layout (bytes); total ~70 MB
#define OFF_SCAL  ((size_t)0)
#define OFF_WQKV  ((size_t)256)
#define OFF_WO    ((size_t)1573120)
#define OFF_W1    ((size_t)2097408)
#define OFF_W3    ((size_t)3539200)
#define OFF_W2    ((size_t)4980992)
#define OFF_XN    ((size_t)6422784)   // xn / yn bf16 (reused)
#define OFF_QKV   ((size_t)14811392)  // qkv bf16; reused for h after attention
#define OFF_VT    ((size_t)39977216)  // V transposed (b,h,d,t) bf16
#define OFF_Y     ((size_t)48365824)  // attention out bf16
#define OFF_Y1    ((size_t)56754432)  // y1 fp32 (residual)

#if defined(__has_builtin)
#if __has_builtin(__builtin_amdgcn_tensor_load_to_lds)
#define HAVE_TDM 1
#endif
#if __has_builtin(__builtin_amdgcn_s_wait_tensorcnt)
#define HAVE_TENSORCNT 1
#endif
#endif

__device__ __forceinline__ bf16_t f2bf(float f) { return (bf16_t)f; }

__device__ __forceinline__ v8f vzero8() {
  v8f v;
#pragma unroll
  for (int i = 0; i < 8; ++i) v[i] = 0.0f;
  return v;
}

// A fragment: 16x32 bf16, lane = row (mod 16).  lanes<16: K = k..k+7, k+16..k+23
//                                               lanes>=16: K = k+8..k+15, k+24..k+31
__device__ __forceinline__ v16bf load_a_frag(const bf16_t* rowp, int k, int hf) {
  v8bf lo = *(const v8bf*)(rowp + k + hf * 8);
  v8bf hi = *(const v8bf*)(rowp + k + 16 + hf * 8);
  v16bf r;
#pragma unroll
  for (int i = 0; i < 8; ++i) { r[i] = lo[i]; r[i + 8] = hi[i]; }
  return r;
}

// B fragment: 32x16 bf16, lane = col (mod 16). lanes<16: K=k..k+7, k+16..k+23; lanes>=16: K=k+8..k+15, k+24..k+31
// colp points to the K-contiguous row of Bt (N x K layout).
__device__ __forceinline__ v16bf load_b_frag(const bf16_t* colp, int k, int hf) {
  v8bf lo = *(const v8bf*)(colp + k + hf * 8);
  v8bf hi = *(const v8bf*)(colp + k + 16 + hf * 8);
  v16bf r;
#pragma unroll
  for (int i = 0; i < 8; ++i) { r[i] = lo[i]; r[i + 8] = hi[i]; }
  return r;
}

#define WMMA_BF16(a, b, c) \
  __builtin_amdgcn_wmma_f32_16x16x32_bf16(false, (a), false, (b), (short)0, (c), false, false)

__device__ __forceinline__ float half_reduce_max(float v) {
#pragma unroll
  for (int o = 1; o < 16; o <<= 1) v = fmaxf(v, __shfl_xor(v, o, 32));
  return v;
}
__device__ __forceinline__ float half_reduce_sum(float v) {
#pragma unroll
  for (int o = 1; o < 16; o <<= 1) v += __shfl_xor(v, o, 32);
  return v;
}

// ---------------------------------------------------------------------------
// TDM: DMA one 32-row x 128-col bf16 tile of Bt (row stride ldK elems) -> LDS.
// Issued by one wave; descriptor per CDNA5 ISA (CDNA5 ISA).
// ---------------------------------------------------------------------------
__device__ __forceinline__ void tdm_stage_b(bf16_t* ldsDst, const bf16_t* gsrc,
                                            int ldK, int lane) {
#ifdef HAVE_TDM
  unsigned lds_off = (unsigned)(unsigned long long)(void*)ldsDst;   // addr[31:0] = LDS offset
  unsigned long long ga = (unsigned long long)gsrc;
  unsigned uK = (unsigned)ldK;
  u32x4 g0;
  g0[0] = 1u;                                        // count=1, user descriptor
  g0[1] = lds_off;                                   // lds_addr [63:32]
  g0[2] = (unsigned)ga;                              // global_addr lo [95:64]
  g0[3] = (unsigned)((ga >> 32) & 0x01FFFFFFu) | 0x80000000u;  // ga hi | type=2 [127:126]
  i32x8 g1;
  g1[0] = (int)(1u << 16);                           // data_size=1 (2 bytes)
  g1[1] = (int)((uK & 0xFFFFu) << 16);               // tensor_dim0 lo -> bits 63:48
  g1[2] = (int)(((uK >> 16) & 0xFFFFu) | (32u << 16)); // tensor_dim0 hi | tensor_dim1=32
  g1[3] = (int)(128u << 16);                         // tile_dim0=128 -> bits 127:112
  g1[4] = (int)32u;                                  // tile_dim1=32, tile_dim2=0
  g1[5] = (int)uK;                                   // tensor_dim0_stride lo
  g1[6] = (int)((uK >> 16) & 0xFFFFu);               // stride hi | dim1_stride lo (0)
  g1[7] = 0;
  i32x4 z4; z4[0] = 0; z4[1] = 0; z4[2] = 0; z4[3] = 0;
#if defined(__clang_major__) && (__clang_major__ >= 23)
  i32x8 z8;
#pragma unroll
  for (int i = 0; i < 8; ++i) z8[i] = 0;
  __builtin_amdgcn_tensor_load_to_lds(g0, g1, z4, z4, z8, 0);
#else
  __builtin_amdgcn_tensor_load_to_lds(g0, g1, z4, z4, 0);
#endif
#else
  // fallback: 32-lane cooperative copy (32 rows x 128 bf16 = 512 x v8bf)
  for (int i = lane; i < 512; i += 32) {
    int row = i >> 4;
    int cb = (i & 15) * 8;
    *(v8bf*)(ldsDst + row * 128 + cb) = *(const v8bf*)(gsrc + (size_t)row * ldK + cb);
  }
#endif
}

// s_wait_tensorcnt needs an immediate operand -> compile-time template constant
template <int N>
__device__ __forceinline__ void tdm_wait() {
#if defined(HAVE_TDM) && defined(HAVE_TENSORCNT)
  __builtin_amdgcn_s_wait_tensorcnt(N);
#endif
}

// ---------------------------------------------------------------------------
// small utility kernels
// ---------------------------------------------------------------------------
__global__ void init_scal_kernel(float* s) {
  if (threadIdx.x < 2) s[threadIdx.x] = 0.0f;
}

// transpose + bf16-convert weights: W (K x N) -> Wt (Npad x Kpad), zero-padded
__global__ void convT_kernel(const float* __restrict__ W, bf16_t* __restrict__ Wt,
                             int K, int N, int Kpad, int Npad) {
  int id = blockIdx.x * 256 + threadIdx.x;
  if (id >= Npad * Kpad) return;
  int n = id / Kpad, k = id % Kpad;
  float v = (n < N && k < K) ? W[(size_t)k * N + n] : 0.0f;
  Wt[(size_t)n * Kpad + k] = f2bf(v);
}

__global__ void sumsq_kernel(const float* __restrict__ p, int n4, float* acc) {
  __shared__ float red[256];
  float s = 0.0f;
  for (int i = blockIdx.x * blockDim.x + threadIdx.x; i < n4; i += gridDim.x * blockDim.x) {
    float4 v = ((const float4*)p)[i];
    s += v.x * v.x + v.y * v.y + v.z * v.z + v.w * v.w;
  }
  red[threadIdx.x] = s;
  __syncthreads();
  for (int o = 128; o > 0; o >>= 1) {
    if (threadIdx.x < o) red[threadIdx.x] += red[threadIdx.x + o];
    __syncthreads();
  }
  if (threadIdx.x == 0) atomicAdd(acc, red[0]);
}

__global__ void rmsnorm_cast_kernel(const float* __restrict__ x, const float* __restrict__ g,
                                    const float* __restrict__ acc, bf16_t* __restrict__ o) {
  int i = blockIdx.x * 256 + threadIdx.x;
  if (i >= TOTALN) return;
  float rms = sqrtf(acc[0] * (1.0f / (float)TOTALN) + 1e-5f);
  o[i] = f2bf(x[i] * (1.0f / rms) * g[i & (EMB - 1)]);
}

// qkv (b,t, h*192 + {q:0..63, k:64..127, v:128..191}) -> vt (b,h,d,t)
__global__ void transpose_v_kernel(const bf16_t* __restrict__ qkv, bf16_t* __restrict__ vt) {
  int id = blockIdx.x * 256 + threadIdx.x;
  if (id >= BATCH * HEADS * HDIM * TSEQ) return;
  int t = id & (TSEQ - 1);
  int d = (id >> 10) & (HDIM - 1);
  int bh = id >> 16;
  int b = bh >> 3, h = bh & 7;
  vt[id] = qkv[(size_t)(b * TSEQ + t) * QKVN + h * 192 + 128 + d];
}

// ---------------------------------------------------------------------------
// WMMA GEMM with TDM-staged B tiles.
// Block = 8 waves covering 256 rows x 32 cols. The 32x K bf16 B-tile is DMAed
// into LDS in 128-element K-chunks (double buffered, TENSORcnt) and shared by
// all waves. A-fragments stream from global (L2 resident).
// EPI==0: store bf16 C.  EPI==1: store f32 C + residual (f32).
// Requires: M % 256 == 0, N % 32 == 0, K % 128 == 0.
// ---------------------------------------------------------------------------
template <int EPI>
__global__ void __launch_bounds__(256) gemm_wmma_kernel(
    const bf16_t* __restrict__ A, const bf16_t* __restrict__ Bt,
    const float* __restrict__ bias, const float* __restrict__ resid,
    void* __restrict__ Cout, int M, int N, int K) {
  __shared__ bf16_t Bs[2][32 * 128];
  int lane = threadIdx.x & 31;
  int ln = lane & 15, hf = lane >> 4;
  int wib = threadIdx.x >> 5;
  int nT2 = N >> 5;
  int mBlk = blockIdx.x / nT2, nT = blockIdx.x % nT2;
  int r0 = mBlk * 256 + wib * 32;
  int c0 = nT * 32;

  const bf16_t* a0p = A + (size_t)(r0 + ln) * K;
  const bf16_t* a1p = a0p + (size_t)16 * K;
  const bf16_t* btile = Bt + (size_t)c0 * K;

  v8f acc[2][2];
  acc[0][0] = vzero8(); acc[0][1] = vzero8();
  acc[1][0] = vzero8(); acc[1][1] = vzero8();

  int nc = K >> 7;   // 128-element K chunks
  if (wib == 0) tdm_stage_b(Bs[0], btile, K, lane);

  for (int c = 0; c < nc; ++c) {
    if (wib == 0) {
      if (c + 1 < nc) {
        tdm_stage_b(Bs[(c + 1) & 1], btile + (size_t)(c + 1) * 128, K, lane);
        tdm_wait<1>();          // chunk c complete (in-order TDM)
      } else {
        tdm_wait<0>();          // drain
      }
    }
    __syncthreads();
    const bf16_t* bs = Bs[c & 1];
#pragma unroll
    for (int kk = 0; kk < 128; kk += 32) {
      int k = c * 128 + kk;
      __builtin_prefetch(a0p + k + 128, 0, 0);
      v16bf a0 = load_a_frag(a0p, k, hf);
      v16bf a1 = load_a_frag(a1p, k, hf);
      v16bf b0 = load_b_frag(bs + ln * 128, kk, hf);
      v16bf b1 = load_b_frag(bs + (16 + ln) * 128, kk, hf);
      acc[0][0] = WMMA_BF16(a0, b0, acc[0][0]);
      acc[0][1] = WMMA_BF16(a0, b1, acc[0][1]);
      acc[1][0] = WMMA_BF16(a1, b0, acc[1][0]);
      acc[1][1] = WMMA_BF16(a1, b1, acc[1][1]);
    }
    __syncthreads();
  }

#pragma unroll
  for (int ti = 0; ti < 2; ++ti) {
#pragma unroll
    for (int tj = 0; tj < 2; ++tj) {
      int n = c0 + tj * 16 + ln;
      int mb = r0 + ti * 16 + hf * 8;
      float bb = bias ? bias[n] : 0.0f;
#pragma unroll
      for (int e = 0; e < 8; ++e) {
        size_t idx = (size_t)(mb + e) * N + n;
        float v = acc[ti][tj][e] + bb;
        if (EPI == 0) ((bf16_t*)Cout)[idx] = f2bf(v);
        else          ((float*)Cout)[idx] = v + resid[idx];
      }
    }
  }
}

// ---------------------------------------------------------------------------
// SwiGLU dual GEMM: H = silu(A*W1t^T + b1) * (A*W3t^T + b3), bf16 out (Npad cols)
// one wave: 16 rows x 32 cols, shared A-fragment, 4 wmma per K-step.
// ---------------------------------------------------------------------------
__global__ void gemm_swiglu_kernel(const bf16_t* __restrict__ A,
                                   const bf16_t* __restrict__ B1t, const bf16_t* __restrict__ B3t,
                                   const float* __restrict__ b1, const float* __restrict__ b3,
                                   bf16_t* __restrict__ H, int M, int N, int Npad, int K) {
  int lane = threadIdx.x & 31;
  int ln = lane & 15, hf = lane >> 4;
  int w = (blockIdx.x << 3) + (threadIdx.x >> 5);
  int nT2 = Npad >> 5;
  int mT = w / nT2, nT = w % nT2;
  int r0 = mT * 16, c0 = nT * 32;

  const bf16_t* ap = A + (size_t)(r0 + ln) * K;
  const bf16_t* p10 = B1t + (size_t)(c0 + ln) * K;
  const bf16_t* p11 = p10 + (size_t)16 * K;
  const bf16_t* p30 = B3t + (size_t)(c0 + ln) * K;
  const bf16_t* p31 = p30 + (size_t)16 * K;

  v8f a1c[2], a3c[2];
  a1c[0] = vzero8(); a1c[1] = vzero8();
  a3c[0] = vzero8(); a3c[1] = vzero8();

  for (int k = 0; k < K; k += 32) {
    v16bf a = load_a_frag(ap, k, hf);
    a1c[0] = WMMA_BF16(a, load_b_frag(p10, k, hf), a1c[0]);
    a1c[1] = WMMA_BF16(a, load_b_frag(p11, k, hf), a1c[1]);
    a3c[0] = WMMA_BF16(a, load_b_frag(p30, k, hf), a3c[0]);
    a3c[1] = WMMA_BF16(a, load_b_frag(p31, k, hf), a3c[1]);
  }

#pragma unroll
  for (int tj = 0; tj < 2; ++tj) {
    int n = c0 + tj * 16 + ln;
    bool valid = (n < N);
    float bb1 = valid ? b1[n] : 0.0f;
    float bb3 = valid ? b3[n] : 0.0f;
#pragma unroll
    for (int e = 0; e < 8; ++e) {
      int row = r0 + hf * 8 + e;
      float v1 = a1c[tj][e] + bb1;
      float v3 = a3c[tj][e] + bb3;
      float hh = valid ? (v1 / (1.0f + __expf(-v1))) * v3 : 0.0f;
      H[(size_t)row * Npad + n] = f2bf(hh);
    }
  }
}

// ---------------------------------------------------------------------------
// Flash-style causal attention. One wave handles (b, h, 16-query tile).
// QK^T and P*V both via WMMA bf16; P converted C-layout -> A-layout through LDS.
// ---------------------------------------------------------------------------
__global__ void attn_kernel(const bf16_t* __restrict__ qkv, const bf16_t* __restrict__ vt,
                            bf16_t* __restrict__ y) {
  __shared__ bf16_t Pbuf[8][16 * 32];
  int lane = threadIdx.x & 31;
  int wib = threadIdx.x >> 5;
  int ln = lane & 15, hf = lane >> 4;
  int w = (blockIdx.x << 3) + wib;      // 0..4095
  int qt = w & 63;
  int bh = w >> 6;
  int b = bh >> 3, h = bh & 7;
  int q0 = qt * 16;

  const bf16_t* Qrow = qkv + (size_t)(b * TSEQ + q0 + ln) * QKVN + h * 192;
  v16bf qa0 = load_a_frag(Qrow, 0, hf);
  v16bf qa1 = load_a_frag(Qrow, 32, hf);

  float rm[8], rs[8];
#pragma unroll
  for (int e = 0; e < 8; ++e) { rm[e] = -1e30f; rs[e] = 0.0f; }
  v8f o0 = vzero8(), o1 = vzero8(), o2 = vzero8(), o3 = vzero8();

  const bf16_t* Vtb = vt + (size_t)bh * HDIM * TSEQ;
  bf16_t* P = Pbuf[wib];

  int kend = q0 + 16;
  for (int kb = 0; kb < kend; kb += 32) {
    // ---- S = Q K^T (two 16-key sub-tiles) ----
    v8f s0 = vzero8(), s1 = vzero8();
    {
      const bf16_t* Kr0 = qkv + (size_t)(b * TSEQ + kb + ln) * QKVN + h * 192 + 64;
      const bf16_t* Kr1 = Kr0 + (size_t)16 * QKVN;
      s0 = WMMA_BF16(qa0, load_b_frag(Kr0, 0, hf), s0);
      s0 = WMMA_BF16(qa1, load_b_frag(Kr0, 32, hf), s0);
      s1 = WMMA_BF16(qa0, load_b_frag(Kr1, 0, hf), s1);
      s1 = WMMA_BF16(qa1, load_b_frag(Kr1, 32, hf), s1);
    }
    // ---- scale + causal mask + online softmax ----
    float corr[8];
#pragma unroll
    for (int e = 0; e < 8; ++e) {
      int qrow = q0 + e + (hf ? 8 : 0);
      float a = s0[e] * ATT_SCALE;
      float c = s1[e] * ATT_SCALE;
      if (kb + ln > qrow) a = -1e30f;
      if (kb + 16 + ln > qrow) c = -1e30f;
      float mx = half_reduce_max(fmaxf(a, c));
      float mn = fmaxf(rm[e], mx);
      corr[e] = __expf(rm[e] - mn);
      rm[e] = mn;
      float p0 = __expf(a - mn);
      float p1 = __expf(c - mn);
      rs[e] = rs[e] * corr[e] + half_reduce_sum(p0 + p1);
      int m = e + (hf ? 8 : 0);
      P[m * 32 + ln] = f2bf(p0);
      P[m * 32 + 16 + ln] = f2bf(p1);
    }
#pragma unroll
    for (int e = 0; e < 8; ++e) {
      o0[e] *= corr[e]; o1[e] *= corr[e]; o2[e] *= corr[e]; o3[e] *= corr[e];
    }
    asm volatile("" ::: "memory");   // order LDS store -> LDS load (same wave, DScnt in-order)
    v16bf pa = load_a_frag(P + ln * 32, 0, hf);
    asm volatile("" ::: "memory");
    // ---- O += P * V ----
    o0 = WMMA_BF16(pa, load_b_frag(Vtb + (size_t)(0 * 16 + ln) * TSEQ, kb, hf), o0);
    o1 = WMMA_BF16(pa, load_b_frag(Vtb + (size_t)(1 * 16 + ln) * TSEQ, kb, hf), o1);
    o2 = WMMA_BF16(pa, load_b_frag(Vtb + (size_t)(2 * 16 + ln) * TSEQ, kb, hf), o2);
    o3 = WMMA_BF16(pa, load_b_frag(Vtb + (size_t)(3 * 16 + ln) * TSEQ, kb, hf), o3);
  }

#pragma unroll
  for (int e = 0; e < 8; ++e) {
    float inv = 1.0f / rs[e];
    int q = q0 + e + (hf ? 8 : 0);
    bf16_t* yp = y + (size_t)(b * TSEQ + q) * EMB + h * HDIM;
    yp[0 + ln]  = f2bf(o0[e] * inv);
    yp[16 + ln] = f2bf(o1[e] * inv);
    yp[32 + ln] = f2bf(o2[e] * inv);
    yp[48 + ln] = f2bf(o3[e] * inv);
  }
}

// ---------------------------------------------------------------------------
// launch
// ---------------------------------------------------------------------------
extern "C" void kernel_launch(void* const* d_in, const int* in_sizes, int n_in,
                              void* d_out, int out_size, void* d_ws, size_t ws_size,
                              hipStream_t stream) {
  const float* x     = (const float*)d_in[0];
  const float* W_w   = (const float*)d_in[1];
  const float* W_b   = (const float*)d_in[2];
  const float* Wo_w  = (const float*)d_in[3];
  const float* Wo_b  = (const float*)d_in[4];
  const float* W1_w  = (const float*)d_in[5];
  const float* W1_b  = (const float*)d_in[6];
  const float* W2_w  = (const float*)d_in[7];
  const float* W2_b  = (const float*)d_in[8];
  const float* W3_w  = (const float*)d_in[9];
  const float* W3_b  = (const float*)d_in[10];
  const float* g_mha = (const float*)d_in[11];
  const float* g_ff  = (const float*)d_in[12];

  char* ws = (char*)d_ws;
  float*  scal  = (float*)(ws + OFF_SCAL);
  bf16_t* Wqkvt = (bf16_t*)(ws + OFF_WQKV);
  bf16_t* Wot   = (bf16_t*)(ws + OFF_WO);
  bf16_t* W1t   = (bf16_t*)(ws + OFF_W1);
  bf16_t* W3t   = (bf16_t*)(ws + OFF_W3);
  bf16_t* W2t   = (bf16_t*)(ws + OFF_W2);
  bf16_t* xn    = (bf16_t*)(ws + OFF_XN);
  bf16_t* qkvb  = (bf16_t*)(ws + OFF_QKV);
  bf16_t* hbuf  = (bf16_t*)(ws + OFF_QKV);   // reuse after attention
  bf16_t* vtb   = (bf16_t*)(ws + OFF_VT);
  bf16_t* ybf   = (bf16_t*)(ws + OFF_Y);
  float*  y1    = (float*)(ws + OFF_Y1);
  float*  out   = (float*)d_out;

  init_scal_kernel<<<1, 64, 0, stream>>>(scal);

  // weight transpose + bf16 convert (N x K layouts)
  convT_kernel<<<(EMB * QKVN + 255) / 256, 256, 0, stream>>>(W_w, Wqkvt, EMB, QKVN, EMB, QKVN);
  convT_kernel<<<(EMB * EMB + 255) / 256, 256, 0, stream>>>(Wo_w, Wot, EMB, EMB, EMB, EMB);
  convT_kernel<<<(EMB * DFFP + 255) / 256, 256, 0, stream>>>(W1_w, W1t, EMB, DFF, EMB, DFFP);
  convT_kernel<<<(EMB * DFFP + 255) / 256, 256, 0, stream>>>(W3_w, W3t, EMB, DFF, EMB, DFFP);
  convT_kernel<<<(DFFP * EMB + 255) / 256, 256, 0, stream>>>(W2_w, W2t, DFF, EMB, DFFP, EMB);

  // rmsnorm(x) -> xn (bf16)
  sumsq_kernel<<<512, 256, 0, stream>>>(x, TOTALN / 4, &scal[0]);
  rmsnorm_cast_kernel<<<TOTALN / 256, 256, 0, stream>>>(x, g_mha, &scal[0], xn);

  // qkv = xn @ W + b   (8192 x 1536), bf16 out
  gemm_wmma_kernel<0><<<(MROWS / 256) * (QKVN / 32), 256, 0, stream>>>(
      xn, Wqkvt, W_b, nullptr, qkvb, MROWS, QKVN, EMB);

  // attention
  transpose_v_kernel<<<TOTALN / 256, 256, 0, stream>>>(qkvb, vtb);
  attn_kernel<<<(BATCH * HEADS * (TSEQ / 16)) / 8, 256, 0, stream>>>(qkvb, vtb, ybf);

  // y1 = y @ Wo + b + x   (fp32)
  gemm_wmma_kernel<1><<<(MROWS / 256) * (EMB / 32), 256, 0, stream>>>(
      ybf, Wot, Wo_b, x, y1, MROWS, EMB, EMB);

  // rmsnorm(y1) -> yn (bf16, reuse xn buffer)
  sumsq_kernel<<<512, 256, 0, stream>>>(y1, TOTALN / 4, &scal[1]);
  rmsnorm_cast_kernel<<<TOTALN / 256, 256, 0, stream>>>(y1, g_ff, &scal[1], xn);

  // h = silu(yn@W1+b1) * (yn@W3+b3)   (bf16, padded to 1408 cols)
  gemm_swiglu_kernel<<<(MROWS / 16) * (DFFP / 32) / 8, 256, 0, stream>>>(
      xn, W1t, W3t, W1_b, W3_b, hbuf, MROWS, DFF, DFFP, EMB);

  // out = y1 + h @ W2 + b2   (fp32)
  gemm_wmma_kernel<1><<<(MROWS / 256) * (EMB / 32), 256, 0, stream>>>(
      hbuf, W2t, W2_b, y1, out, MROWS, EMB, DFFP);
}
